// AnteLayer_56014963474584
// MI455X (gfx1250) — matrix operand order repaired and checked by reading, not compiled, not run on hardware
//
#include <hip/hip_runtime.h>
#include <hip/hip_bf16.h>
#include <stdint.h>

typedef __attribute__((ext_vector_type(16))) _Float16       v16h;
typedef __attribute__((ext_vector_type(16))) unsigned short v16u;
typedef __attribute__((ext_vector_type(8)))  float          v8f;

union V16U { v16u v; v16h h; uint32_t u[8]; };
union V8F  { v8f  v; float f[8]; };

static __device__ __forceinline__ uint32_t pack2h(float a, float b) {
    union { _Float16 f; uint16_t s; } lo, hi;
    lo.f = (_Float16)a; hi.f = (_Float16)b;
    return ((uint32_t)hi.s << 16) | (uint32_t)lo.s;
}

#define NCHUNK 6   // 6 * 32 = 192 >= 161 z-points (padded with zero membership)
#define NRULE  5

__global__ __launch_bounds__(256) void fuzzy_edge_kernel(
    const float* __restrict__ hnode,
    const int*   __restrict__ src_idx,
    const int*   __restrict__ dst_idx,
    float*       __restrict__ out,
    int E, int ntiles)
{
    // packed-half zmf table, WMMA-A-friendly layout: [chunk][rule][16 x half2]
    __shared__ uint32_t zmfT[NCHUNK * NRULE * 16];   // 1920 B
    __shared__ float    stage[8][32];                // per-wave C extraction

    // ---- build consequent-membership table once per block -----------------
    for (int t = threadIdx.x; t < NCHUNK * NRULE * 16; t += blockDim.x) {
        int chunk = t / (NRULE * 16);
        int rule  = (t / 16) % NRULE;
        int u     = t % 16;
        int k0 = chunk * 32 + 2 * u;
        float c  = -0.3f + 0.4f * (float)rule;           // OUT_C[rule]
        float z0 = -0.3f + 0.01f * (float)k0;
        float z1 = z0 + 0.01f;
        float d0 = (z0 - c) * (1.0f / 0.3f);
        float d1 = (z1 - c) * (1.0f / 0.3f);
        float g0 = (k0     < 161) ? __expf(-0.5f * d0 * d0) : 0.0f;
        float g1 = (k0 + 1 < 161) ? __expf(-0.5f * d1 * d1) : 0.0f;
        zmfT[t] = pack2h(g0, g1);
    }
    __syncthreads();

    const int  lane   = threadIdx.x & 31;
    const int  wave   = threadIdx.x >> 5;
    const int  tile   = blockIdx.x * 8 + wave;
    if (tile >= ntiles) return;                          // wave-uniform exit

    const int  m      = lane & 15;        // edge row: lanes l and l+16 share it
    const bool hiHalf = lane >= 16;
    const int  e      = tile * 16 + m;
    const int  eld    = (e < E) ? e : 0;

    // ---- phase 1: antecedents (gather, norm, angle, memberships) ----------
    int si = src_idx[eld], di = dst_idx[eld];
    const float4* h4 = (const float4*)hnode;
    float4 da = h4[di * 2 + 0], db = h4[di * 2 + 1];
    float4 sa = h4[si * 2 + 0], sb = h4[si * 2 + 1];
    float v0 = da.x - sa.x, v1 = da.y - sa.y, v2 = da.z - sa.z, v3 = da.w - sa.w;
    float v4 = db.x - sb.x, v5 = db.y - sb.y, v6 = db.z - sb.z, v7 = db.w - sb.w;
    float n2 = v0*v0 + v1*v1 + v2*v2 + v3*v3 + v4*v4 + v5*v5 + v6*v6 + v7*v7 + 1e-12f;
    float nrm = __builtin_sqrtf(n2);
    float x1  = fminf(nrm, 4.0f);
    float ca  = fminf(fmaxf(v0 / nrm, -0.999999f), 0.999999f);
    float x2  = acosf(ca) * 57.29577951308232f;          // degrees

    float mu1[5], mu2[5];
    #pragma unroll
    for (int i = 0; i < 5; ++i) {
        float a = x1 - (float)i;
        mu1[i] = __expf(-0.5f * a * a);
        float b = (x2 - 45.0f * (float)i) * (1.0f / 45.0f);
        mu2[i] = __expf(-0.5f * b * b);
    }

    // ---- Mamdani rule activations (max over OR-terms of min) --------------
    float act[5];
    {
        float a;
        a = fminf(mu1[2], mu2[4]);
        a = fmaxf(a, fminf(mu1[3], mu2[4]));
        a = fmaxf(a, fminf(mu1[3], mu2[3]));
        a = fmaxf(a, fminf(mu1[4], mu2[3]));
        a = fmaxf(a, fminf(mu1[4], mu2[4]));
        act[0] = a;
        a = fminf(mu1[1], mu2[4]);
        a = fmaxf(a, fminf(mu1[2], mu2[3]));
        a = fmaxf(a, fminf(mu1[3], mu2[2]));
        a = fmaxf(a, fminf(mu1[4], mu2[1]));
        act[1] = a;
        a = fminf(mu1[0], mu2[4]);
        a = fmaxf(a, fminf(mu1[1], mu2[3]));
        a = fmaxf(a, fminf(mu1[2], mu2[2]));
        a = fmaxf(a, fminf(mu1[3], mu2[1]));
        a = fmaxf(a, fminf(mu1[4], mu2[0]));
        act[2] = a;
        a = fminf(mu1[0], mu2[3]);
        a = fmaxf(a, fminf(mu1[1], mu2[2]));
        a = fmaxf(a, fminf(mu1[2], mu2[1]));
        a = fmaxf(a, fminf(mu1[3], mu2[0]));
        act[3] = a;
        a = fminf(mu1[0], mu2[2]);
        a = fmaxf(a, fminf(mu1[0], mu2[1]));
        a = fmaxf(a, fminf(mu1[0], mu2[0]));
        a = fmaxf(a, fminf(mu1[1], mu2[1]));
        a = fmaxf(a, fminf(mu1[1], mu2[0]));
        a = fmaxf(a, fminf(mu1[2], mu2[0]));
        act[4] = a;
    }
    if (e >= E) { act[0] = act[1] = act[2] = act[3] = act[4] = 0.0f; }

    // splat activations to packed vectors (u16 domain: data is nonneg f16,
    // so unsigned-int compare == IEEE compare, no canonicalization needed)
    v16u actv[5];
    #pragma unroll
    for (int r = 0; r < NRULE; ++r) {
        V16U t; uint32_t p = pack2h(act[r], act[r]);
        #pragma unroll
        for (int q = 0; q < 8; ++q) t.u[q] = p;
        actv[r] = t.v;
    }

    // ---- B fragment (32x16 f16): col 0 = ones, col 1 = z, rest 0 ----------
    // lanes 0-15: N=lane, K=0..7,16..23; lanes 16-31: N=lane-16, K=8..15,24..31 (vgpr q -> 2 consecutive K)
    V16U b;  V16U bs;
    {
        float one = (m == 0) ? 1.0f : 0.0f;
        bool  isZ = (m == 1);
        #pragma unroll
        for (int q = 0; q < 8; ++q) {
            int k = 2 * q + (q >= 4 ? 8 : 0) + (hiHalf ? 8 : 0);
            float z0 = -0.3f + 0.01f * (float)k;
            b.u[q] = pack2h(isZ ? z0 : one, isZ ? z0 + 0.01f : one);
        }
        float dz = isZ ? 0.32f : 0.0f;                   // 32 z-steps per chunk
        uint32_t ps = pack2h(dz, dz);
        #pragma unroll
        for (int q = 0; q < 8; ++q) bs.u[q] = ps;
    }

    // ---- phase 2: chunked implication + WMMA weighted reduction -----------
    v8f c = {};
    const int selBase = hiHalf ? 4 : 0;                  // A-layout K-half select
    #pragma unroll
    for (int chunk = 0; chunk < NCHUNK; ++chunk) {
        V16U agg;
        #pragma unroll
        for (int r = 0; r < NRULE; ++r) {
            int base = (chunk * NRULE + r) * 16 + selBase;
            uint4 lo4 = *(const uint4*)&zmfT[base];       // ds_load_b128
            uint4 hi4 = *(const uint4*)&zmfT[base + 8];   // ds_load_b128
            V16U zm;
            zm.u[0] = lo4.x; zm.u[1] = lo4.y; zm.u[2] = lo4.z; zm.u[3] = lo4.w;
            zm.u[4] = hi4.x; zm.u[5] = hi4.y; zm.u[6] = hi4.z; zm.u[7] = hi4.w;
            v16u t = __builtin_elementwise_min(zm.v, actv[r]);   // v_pk_min_u16
            agg.v = (r == 0) ? t : __builtin_elementwise_max(agg.v, t); // v_pk_max_u16
        }
        // C += A(16x32 agg) x B(32x16 [1|z]) : per-edge [Σagg, Σagg*z]
        c = __builtin_amdgcn_wmma_f32_16x16x32_f16(false, agg.h, false, b.h,
                                                   (short)0, c, false, false);
        if (chunk + 1 < NCHUNK) b.h = b.h + bs.h;        // v_pk_add_f16: z += 0.32
    }

    // ---- extract C and defuzzify ------------------------------------------
    // C[M][N]: M<8 -> vgpr M / lane N ; M>=8 -> vgpr M-8 / lane 16+N
    // lane 0: den edges 0-7 | lane 16: den edges 8-15 |
    // lane 1: num edges 0-7 | lane 17: num edges 8-15
    V8F cu; cu.v = c;
    {
        int slot = -1;
        if (lane == 0)  slot = 0;
        if (lane == 16) slot = 8;
        if (lane == 1)  slot = 16;
        if (lane == 17) slot = 24;
        if (slot >= 0) {
            #pragma unroll
            for (int g = 0; g < 8; ++g) stage[wave][slot + g] = cu.f[g];
        }
    }
    asm volatile("s_wait_dscnt 0" ::: "memory");         // in-wave LDS RAW fence
    __builtin_amdgcn_wave_barrier();
    if (lane < 16 && e < E) {
        float den = stage[wave][m];
        float num = stage[wave][16 + m];
        out[e] = num / den;
    }
}

extern "C" void kernel_launch(void* const* d_in, const int* in_sizes, int n_in,
                              void* d_out, int out_size, void* d_ws, size_t ws_size,
                              hipStream_t stream) {
    const float* h   = (const float*)d_in[0];
    const int*   src = (const int*)d_in[1];
    const int*   dst = (const int*)d_in[2];
    // d_in[3] = etypes: unused by the reference computation
    float* out = (float*)d_out;
    int E = in_sizes[1];
    int ntiles = (E + 15) / 16;          // 16 edges per wave tile
    int blocks = (ntiles + 7) / 8;       // 8 waves per 256-thread block
    fuzzy_edge_kernel<<<blocks, 256, 0, stream>>>(h, src, dst, out, E, ntiles);
}
